// FPN_Net_69655779607245
// MI455X (gfx1250) — matrix operand order repaired and checked by reading, not compiled, not run on hardware
//
#include <hip/hip_runtime.h>
#include <cstdint>
#include <cstddef>

// ---------------------------------------------------------------------------
// CDNA5 (gfx1250) sparse FPN U-Net, WMMA bf16 implicit-GEMM implementation.
// wave32; matrix ops via v_wmma_f32_16x16x32_bf16.
// - activations bf16 channels-last [voxel][C] (~136MB total -> L2-resident)
// - each wave: 16 voxels x 32 cout (two accumulators, shared A fragment)
// - (Cin,Cout,DIM) compile-time: tap offsets fold into load immediates
// - boundary taps: branchless clamp to a zeroed guard page (no EXEC churn)
// ---------------------------------------------------------------------------

typedef __bf16 bf16_t;
typedef __attribute__((ext_vector_type(16))) __bf16 v16bf_t;
typedef __attribute__((ext_vector_type(8)))  __bf16 v8bf_t;
typedef __attribute__((ext_vector_type(8)))  float  v8f_t;

#define BN_EPS 1e-4f

union AFrag { v16bf_t v; v8bf_t h[2]; bf16_t e[16]; };

__device__ __forceinline__ v8f_t wmma_bf16(v16bf_t a, v16bf_t b, v8f_t c) {
  return __builtin_amdgcn_wmma_f32_16x16x32_bf16(false, a, false, b, (short)0, c,
                                                 false, false);
}

template <int DIM> struct Log2 { static constexpr int v = 1 + Log2<DIM / 2>::v; };
template <> struct Log2<1> { static constexpr int v = 0; };

// ------------------------------- masks -------------------------------------
__global__ void k_mask0(const unsigned char* __restrict__ m, float* __restrict__ mf, int n) {
  int i = blockIdx.x * blockDim.x + threadIdx.x;
  if (i < n) mf[i] = m[i] ? 1.0f : 0.0f;
}

__global__ void k_pool(const float* __restrict__ src, float* __restrict__ dst, int dc, int NV) {
  int i = blockIdx.x * blockDim.x + threadIdx.x;
  if (i >= NV) return;
  int Sc = dc * dc * dc, df = dc * 2;
  int b = i / Sc, s = i % Sc;
  int d = s / (dc * dc), h = (s / dc) % dc, w = s % dc;
  float any = 0.0f;
#pragma unroll
  for (int t = 0; t < 8; ++t) {
    int p = (t >> 2) & 1, q = (t >> 1) & 1, r = t & 1;
    any = fmaxf(any, src[(((size_t)b * df + 2 * d + p) * df + 2 * h + q) * df + 2 * w + r]);
  }
  dst[i] = any > 0.0f ? 1.0f : 0.0f;
}

__global__ void k_count(const float* __restrict__ mf, float* __restrict__ cnt, int n) {
  int i = blockIdx.x * blockDim.x + threadIdx.x;
  int lane = threadIdx.x & 31;
  int act = (i < n && mf[i] > 0.0f) ? 1 : 0;
  unsigned long long bal = __ballot(act);
  if (lane == 0) atomicAdd(cnt, (float)__popcll(bal));
}

// --------------------------- weight repacking ------------------------------
// dst fragment layout: (((t*KC + kc)*CT + ct) << 9) + lane*16 + h
// B[k][n] with co = ct*16 + lane%16 ; k = (lane/16)*8 + (h&7) + (h>>3)*16 ; ci = kc*32 + k
__global__ void k_repack(const float* __restrict__ W, bf16_t* __restrict__ dst,
                         int T, int Cin, int Cout, int total) {
  int i = blockIdx.x * blockDim.x + threadIdx.x;
  if (i >= total) return;
  int hh = i & 15;
  int lane = (i >> 4) & 31;
  int rest = i >> 9;
  int CT = Cout >> 4, KC = Cin >> 5;
  int ct = rest % CT; rest /= CT;
  int kc = rest % KC;
  int t = rest / KC;
  int co = ct * 16 + (lane & 15);
  int k = ((lane >> 4) << 3) + (hh & 7) + ((hh >> 3) << 4);
  int ci = kc * 32 + k;
  dst[i] = (bf16_t)W[((size_t)co * Cin + ci) * T + t];
}

// ------------------------------ input conv ---------------------------------
__global__ void k_inconv(const float* __restrict__ x, const float* __restrict__ mf,
                         const float* __restrict__ Win, bf16_t* __restrict__ out,
                         int dim, int NV) {
  int idx = blockIdx.x * blockDim.x + threadIdx.x;
  if (idx >= NV * 4) return;
  int v = idx >> 2;
  int cg = (idx & 3) * 8;
  int S = dim * dim * dim;
  if (mf[v] == 0.0f) {
#pragma unroll
    for (int j = 0; j < 8; ++j) out[(size_t)v * 32 + cg + j] = (bf16_t)0.0f;
    return;
  }
  int b = v / S, s = v % S;
  int d = s / (dim * dim), h = (s / dim) % dim, w = s % dim;
  float acc[8];
#pragma unroll
  for (int j = 0; j < 8; ++j) acc[j] = 0.0f;
  for (int t = 0; t < 27; ++t) {
    int nd = d + t / 9 - 1, nh = h + (t / 3) % 3 - 1, nw = w + t % 3 - 1;
    if (nd < 0 || nd >= dim || nh < 0 || nh >= dim || nw < 0 || nw >= dim) continue;
    int ns = (nd * dim + nh) * dim + nw;
    float gm = mf[(size_t)b * S + ns];
    if (gm == 0.0f) continue;
    float x0 = x[((size_t)b * 3 + 0) * S + ns];
    float x1 = x[((size_t)b * 3 + 1) * S + ns];
    float x2 = x[((size_t)b * 3 + 2) * S + ns];
#pragma unroll
    for (int j = 0; j < 8; ++j) {
      const float* wp = Win + ((size_t)(cg + j) * 3) * 27 + t;
      acc[j] += x0 * wp[0] + x1 * wp[27] + x2 * wp[54];
    }
  }
#pragma unroll
  for (int j = 0; j < 8; ++j) out[(size_t)v * 32 + cg + j] = (bf16_t)acc[j];
}

// ----------------------------- batch-norm ----------------------------------
__global__ void k_bnstats(const bf16_t* __restrict__ in, const float* __restrict__ mf,
                          int NV, int C, float* __restrict__ stats) {
  __shared__ float ls[128];
  __shared__ float lq[128];
  int tid = threadIdx.x;
  if (tid < C) { ls[tid] = 0.0f; lq[tid] = 0.0f; }
  __syncthreads();
  int v0 = blockIdx.x * 64;
  for (int i = tid; i < 64 * C; i += blockDim.x) {
    int v = v0 + i / C;
    int c = i % C;
    if (mf[v] > 0.0f) {
      float xv = (float)in[(size_t)v * C + c];
      atomicAdd(&ls[c], xv);
      atomicAdd(&lq[c], xv * xv);
    }
  }
  __syncthreads();
  if (tid < C) {
    atomicAdd(&stats[tid], ls[tid]);
    atomicAdd(&stats[128 + tid], lq[tid]);
  }
}

__global__ void k_bnapply(const bf16_t* __restrict__ in, bf16_t* __restrict__ out,
                          const float* __restrict__ mf, const float* __restrict__ g,
                          const float* __restrict__ bb, const float* __restrict__ stats,
                          const float* __restrict__ cnt, int NV, int C) {
  int i = blockIdx.x * blockDim.x + threadIdx.x;
  if (i >= NV * C) return;
  int v = i / C, c = i % C;
  float m = mf[v];
  float n = *cnt;
  float mu = stats[c] / n;
  float var = stats[128 + c] / n - mu * mu;
  float y = ((float)in[i] - mu) * rsqrtf(var + BN_EPS) * g[c] + bb[c];
  y = y > 0.0f ? y : 0.0f;   // leakiness = 0
  out[i] = (bf16_t)(y * m);
}

// ---------------------- submanifold 3x3x3 conv (WMMA) ----------------------
// One wave: 16 voxels x 32 cout (two 16x16 tiles, shared A, dual accumulators).
// Boundary taps read from a zeroed guard page via branchless pointer select.
template <int Cin, int Cout, int DIM>
__global__ void k_subm3(const bf16_t* __restrict__ in, bf16_t* __restrict__ out,
                        const float* __restrict__ mf, const bf16_t* __restrict__ wf,
                        const bf16_t* __restrict__ zp) {
  constexpr int KC = Cin / 32;
  constexpr int NG = Cout / 32;
  constexpr int CT = Cout / 16;
  constexpr int TT = 27 * KC;
  constexpr int L = Log2<DIM>::v;
  constexpr int S = DIM * DIM * DIM;
  constexpr int NV = 2 * S;

  const int lane = threadIdx.x & 31;
  const int wid = blockIdx.x * (blockDim.x >> 5) + (threadIdx.x >> 5);
  if (wid >= (NV >> 4) * NG) return;
  const int g = wid % NG;
  const int vb = (wid / NG) << 4;
  const int v = vb + (lane & 15);
  const int s = v & (S - 1);
  const int d = (s >> (2 * L)) & (DIM - 1);
  const int h = (s >> L) & (DIM - 1);
  const int w = s & (DIM - 1);
  const int hi8 = (lane >> 4) << 3;   // 0 or 8: A-fragment K-half / C-row half
  const int co0 = g * 32 + (lane & 15);

  // 3-bit bounds predicates; bit i tested for delta = i-1; combined per tap.
  const unsigned okd = (unsigned)(d > 0) | 2u | ((unsigned)(d < DIM - 1) << 2);
  const unsigned okh = (unsigned)(h > 0) | 2u | ((unsigned)(h < DIM - 1) << 2);
  const unsigned okw = (unsigned)(w > 0) | 2u | ((unsigned)(w < DIM - 1) << 2);
  unsigned okbits = 0;
#pragma unroll
  for (int t = 0; t < 27; ++t) {
    const int dz = t / 9, dy = (t / 3) % 3, dx = t % 3;
    okbits |= ((((okd >> dz) & (okh >> dy) & (okw >> dx)) & 1u) << t);
  }

  bf16_t* outRow = out + (size_t)(vb + hi8) * Cout + co0;
  const float* mRow = mf + vb + hi8;

  unsigned long long bal = __ballot((mf[v] > 0.0f) && (lane < 16));
  if (bal == 0ull) {
#pragma unroll
    for (int r = 0; r < 8; ++r) {
      outRow[(size_t)r * Cout] = (bf16_t)0.0f;
      outRow[(size_t)r * Cout + 16] = (bf16_t)0.0f;
    }
    return;
  }

  const bf16_t* inV = in + (size_t)v * Cin + hi8;                // A base (per lane)
  const bf16_t* wfB = wf + ((size_t)(g * 2) << 9) + lane * 16;   // B base (per lane)

  v8f_t c0, c1;
#pragma unroll
  for (int j = 0; j < 8; ++j) { c0[j] = 0.0f; c1[j] = 0.0f; }

  AFrag a[2], b0[2], b1[2];
  auto stage = [&](int slot, int it) {
    const int t = it / KC, kc = it % KC;
    const int dz = t / 9, dy = (t / 3) % 3, dx = t % 3;  // 0..2
    const int tofs = ((dz - 1) * DIM * DIM + (dy - 1) * DIM + (dx - 1)) * Cin;
    // Branchless: out-of-bounds taps read the zeroed guard page.
    const bf16_t* pt = ((okbits >> t) & 1u) ? (inV + tofs) : zp;  // CSE across kc
    a[slot].h[0] = *(const v8bf_t*)(pt + kc * 32);
    a[slot].h[1] = *(const v8bf_t*)(pt + kc * 32 + 16);
    const int bofs = ((t * KC + kc) * CT) << 9;
    b0[slot].h[0] = *(const v8bf_t*)(wfB + bofs);
    b0[slot].h[1] = *(const v8bf_t*)(wfB + bofs + 8);
    b1[slot].h[0] = *(const v8bf_t*)(wfB + bofs + 512);
    b1[slot].h[1] = *(const v8bf_t*)(wfB + bofs + 520);
  };

  stage(0, 0);
#pragma unroll
  for (int it = 0; it < TT; ++it) {
    const int cur = it & 1;
    if (it + 1 < TT) stage(cur ^ 1, it + 1);
    c0 = wmma_bf16(a[cur].v, b0[cur].v, c0);
    c1 = wmma_bf16(a[cur].v, b1[cur].v, c1);
  }

#pragma unroll
  for (int r = 0; r < 8; ++r) {
    float m = mRow[r];
    outRow[(size_t)r * Cout] = (bf16_t)(c0[r] * m);
    outRow[(size_t)r * Cout + 16] = (bf16_t)(c1[r] * m);
  }
}

// ------------------------ 2x2x2 stride-2 down-conv -------------------------
template <int Cin, int Cout, int DIMC>
__global__ void k_down(const bf16_t* __restrict__ in, bf16_t* __restrict__ out,
                       const float* __restrict__ mfc, const bf16_t* __restrict__ wf) {
  constexpr int KC = Cin / 32;
  constexpr int NG = Cout / 32;
  constexpr int CT = Cout / 16;
  constexpr int TT = 8 * KC;
  constexpr int L = Log2<DIMC>::v;
  constexpr int Sc = DIMC * DIMC * DIMC;
  constexpr int NVc = 2 * Sc;
  constexpr int DF = 2 * DIMC;

  const int lane = threadIdx.x & 31;
  const int wid = blockIdx.x * (blockDim.x >> 5) + (threadIdx.x >> 5);
  if (wid >= (NVc >> 4) * NG) return;
  const int g = wid % NG;
  const int vb = (wid / NG) << 4;
  const int v = vb + (lane & 15);
  const int b = v >> (3 * L);
  const int s = v & (Sc - 1);
  const int d = (s >> (2 * L)) & (DIMC - 1);
  const int h = (s >> L) & (DIMC - 1);
  const int w = s & (DIMC - 1);
  const int hi8 = (lane >> 4) << 3;
  const int co0 = g * 32 + (lane & 15);

  bf16_t* outRow = out + (size_t)(vb + hi8) * Cout + co0;
  const float* mRow = mfc + vb + hi8;

  unsigned long long bal = __ballot((mfc[v] > 0.0f) && (lane < 16));
  if (bal == 0ull) {
#pragma unroll
    for (int r = 0; r < 8; ++r) {
      outRow[(size_t)r * Cout] = (bf16_t)0.0f;
      outRow[(size_t)r * Cout + 16] = (bf16_t)0.0f;
    }
    return;
  }

  const int fbase = ((b * DF + 2 * d) * DF + 2 * h) * DF + 2 * w;
  const bf16_t* inF = in + (size_t)fbase * Cin + hi8;
  const bf16_t* wfB = wf + ((size_t)(g * 2) << 9) + lane * 16;

  v8f_t c0, c1;
#pragma unroll
  for (int j = 0; j < 8; ++j) { c0[j] = 0.0f; c1[j] = 0.0f; }

  AFrag a[2], b0[2], b1[2];
  auto stage = [&](int slot, int it) {
    const int t = it / KC, kc = it % KC;
    const int p = (t >> 2) & 1, q = (t >> 1) & 1, r = t & 1;
    const int aofs = (p * DF * DF + q * DF + r) * Cin + kc * 32;
    a[slot].h[0] = *(const v8bf_t*)(inF + aofs);
    a[slot].h[1] = *(const v8bf_t*)(inF + aofs + 16);
    const int bofs = ((t * KC + kc) * CT) << 9;
    b0[slot].h[0] = *(const v8bf_t*)(wfB + bofs);
    b0[slot].h[1] = *(const v8bf_t*)(wfB + bofs + 8);
    b1[slot].h[0] = *(const v8bf_t*)(wfB + bofs + 512);
    b1[slot].h[1] = *(const v8bf_t*)(wfB + bofs + 520);
  };

  stage(0, 0);
#pragma unroll
  for (int it = 0; it < TT; ++it) {
    const int cur = it & 1;
    if (it + 1 < TT) stage(cur ^ 1, it + 1);
    c0 = wmma_bf16(a[cur].v, b0[cur].v, c0);
    c1 = wmma_bf16(a[cur].v, b1[cur].v, c1);
  }

#pragma unroll
  for (int r = 0; r < 8; ++r) {
    float m = mRow[r];
    outRow[(size_t)r * Cout] = (bf16_t)(c0[r] * m);
    outRow[(size_t)r * Cout + 16] = (bf16_t)(c1[r] * m);
  }
}

// ------------------------ 2x2x2 stride-2 deconv ----------------------------
// Cin = Cout = 32. One wave: 16 coarse voxels x 32 cout, 8 parity taps.
template <int DIMC>
__global__ void k_deconv(const bf16_t* __restrict__ in, bf16_t* __restrict__ out,
                         const float* __restrict__ mff, const bf16_t* __restrict__ wf) {
  constexpr int L = Log2<DIMC>::v;
  constexpr int Sc = DIMC * DIMC * DIMC;
  constexpr int NVc = 2 * Sc;
  constexpr int DF = 2 * DIMC;

  const int lane = threadIdx.x & 31;
  const int wid = blockIdx.x * (blockDim.x >> 5) + (threadIdx.x >> 5);
  if (wid >= (NVc >> 4)) return;
  const int vb = wid << 4;
  const int hi8 = (lane >> 4) << 3;
  const int co0 = lane & 15;

  const bf16_t* ap = in + (size_t)(vb + (lane & 15)) * 32 + hi8;
  AFrag a;
  a.h[0] = *(const v8bf_t*)ap;
  a.h[1] = *(const v8bf_t*)(ap + 16);

  // Fine-grid linear base per output row (coarse voxel): all tap offsets const.
  int vfbase[8];
#pragma unroll
  for (int r = 0; r < 8; ++r) {
    int vs = vb + r + hi8;
    int b = vs >> (3 * L);
    int sc = vs & (Sc - 1);
    int d = (sc >> (2 * L)) & (DIMC - 1);
    int h = (sc >> L) & (DIMC - 1);
    int w = sc & (DIMC - 1);
    vfbase[r] = ((b * DF + 2 * d) * DF + 2 * h) * DF + 2 * w;
  }

  const bf16_t* wfB = wf + lane * 16;
#pragma unroll
  for (int t = 0; t < 8; ++t) {
    AFrag bf0, bf1;
    const int bofs = (t * 2) << 9;
    bf0.h[0] = *(const v8bf_t*)(wfB + bofs);
    bf0.h[1] = *(const v8bf_t*)(wfB + bofs + 8);
    bf1.h[0] = *(const v8bf_t*)(wfB + bofs + 512);
    bf1.h[1] = *(const v8bf_t*)(wfB + bofs + 520);
    v8f_t c0, c1;
#pragma unroll
    for (int j = 0; j < 8; ++j) { c0[j] = 0.0f; c1[j] = 0.0f; }
    c0 = wmma_bf16(a.v, bf0.v, c0);
    c1 = wmma_bf16(a.v, bf1.v, c1);
    const int p = (t >> 2) & 1, q = (t >> 1) & 1, r2 = t & 1;
    const int fofs = p * DF * DF + q * DF + r2;  // compile-time per tap
#pragma unroll
    for (int r = 0; r < 8; ++r) {
      float m = mff[vfbase[r] + fofs];
      bf16_t* po = out + (((size_t)vfbase[r] + fofs) << 5) + co0;
      po[0] = (bf16_t)(c0[r] * m);
      po[16] = (bf16_t)(c1[r] * m);
    }
  }
}

// -------------------------- 1x1 shortcut (WMMA) ----------------------------
template <int Cin>
__global__ void k_shortcut(const bf16_t* __restrict__ in, bf16_t* __restrict__ out,
                           const float* __restrict__ mf, const bf16_t* __restrict__ wf,
                           int NV, int beta) {
  constexpr int KC = Cin / 32;
  const int lane = threadIdx.x & 31;
  const int wid = blockIdx.x * (blockDim.x >> 5) + (threadIdx.x >> 5);
  if (wid >= (NV >> 4)) return;
  const int vb = wid << 4;
  const int hi8 = (lane >> 4) << 3;
  const int co0 = lane & 15;

  v8f_t c0, c1;
#pragma unroll
  for (int j = 0; j < 8; ++j) { c0[j] = 0.0f; c1[j] = 0.0f; }

  const bf16_t* ap = in + (size_t)(vb + (lane & 15)) * Cin + hi8;
  const bf16_t* wfB = wf + lane * 16;
#pragma unroll
  for (int kc = 0; kc < KC; ++kc) {
    AFrag a;
    a.h[0] = *(const v8bf_t*)(ap + kc * 32);
    a.h[1] = *(const v8bf_t*)(ap + kc * 32 + 16);
    AFrag bf0, bf1;
    const int bofs = (kc * 2) << 9;
    bf0.h[0] = *(const v8bf_t*)(wfB + bofs);
    bf0.h[1] = *(const v8bf_t*)(wfB + bofs + 8);
    bf1.h[0] = *(const v8bf_t*)(wfB + bofs + 512);
    bf1.h[1] = *(const v8bf_t*)(wfB + bofs + 520);
    c0 = wmma_bf16(a.v, bf0.v, c0);
    c1 = wmma_bf16(a.v, bf1.v, c1);
  }
  bf16_t* outRow = out + (size_t)(vb + hi8) * 32 + co0;
  const float* mRow = mf + vb + hi8;
#pragma unroll
  for (int r = 0; r < 8; ++r) {
    float m = mRow[r];
    float o0 = beta ? (float)outRow[(size_t)r * 32] : 0.0f;
    float o1 = beta ? (float)outRow[(size_t)r * 32 + 16] : 0.0f;
    outRow[(size_t)r * 32] = (bf16_t)((o0 + c0[r]) * m);
    outRow[(size_t)r * 32 + 16] = (bf16_t)((o1 + c1[r]) * m);
  }
}

// ------------------------------- export ------------------------------------
__global__ void k_export(const bf16_t* __restrict__ in, float* __restrict__ outp,
                         int dim, int NV) {
  int i = blockIdx.x * blockDim.x + threadIdx.x;
  if (i >= NV * 32) return;
  int v = i >> 5, c = i & 31;
  int S = dim * dim * dim;
  int b = v / S, s = v % S;
  outp[((size_t)(b * 32 + c)) * S + s] = (float)in[i];
}

// ---------------------------------------------------------------------------
extern "C" void kernel_launch(void* const* d_in, const int* in_sizes, int n_in,
                              void* d_out, int out_size, void* d_ws, size_t ws_size,
                              hipStream_t stream) {
  (void)in_sizes; (void)out_size;
  if (n_in < 40) return;

  const int DIMS[4] = {64, 32, 16, 8};
  const int NV[4] = {2 * 64 * 64 * 64, 2 * 32 * 32 * 32, 2 * 16 * 16 * 16, 2 * 8 * 8 * 8};

  // ---- workspace arena ----
  uint8_t* base = (uint8_t*)d_ws;
  size_t off = 0;
  auto alloc = [&](size_t bytes) -> void* {
    off = (off + 255) & ~(size_t)255;
    void* p = base + off;
    off += bytes;
    return p;
  };
  float* mf[4];
  for (int k = 0; k < 4; ++k) mf[k] = (float*)alloc((size_t)NV[k] * 4);
  float* cnt = (float*)alloc(4 * 4);
  float* stats = (float*)alloc(256 * 4);
  bf16_t* zp = (bf16_t*)alloc(512);   // zeroed guard page for boundary taps

  auto wsz = [](int T, int Cin, int Cout) -> size_t {
    return (size_t)T * (Cin >> 5) * (Cout >> 4) * 512 * sizeof(bf16_t);
  };
  bf16_t* wb[4];
  wb[0] = (bf16_t*)alloc(wsz(27, 32, 32));
  wb[1] = (bf16_t*)alloc(wsz(27, 64, 64));
  wb[2] = (bf16_t*)alloc(wsz(27, 96, 96));
  wb[3] = (bf16_t*)alloc(wsz(27, 128, 128));
  bf16_t* wd[4];
  wd[1] = (bf16_t*)alloc(wsz(8, 32, 64));
  wd[2] = (bf16_t*)alloc(wsz(8, 64, 96));
  wd[3] = (bf16_t*)alloc(wsz(8, 96, 128));
  bf16_t* wsc[4];
  wsc[0] = (bf16_t*)alloc(wsz(1, 32, 32));
  wsc[1] = (bf16_t*)alloc(wsz(1, 64, 32));
  wsc[2] = (bf16_t*)alloc(wsz(1, 96, 32));
  wsc[3] = (bf16_t*)alloc(wsz(1, 128, 32));
  bf16_t *wu[3], *wm[3];
  for (int k = 0; k < 3; ++k) {
    wu[k] = (bf16_t*)alloc(wsz(8, 32, 32));
    wm[k] = (bf16_t*)alloc(wsz(27, 32, 32));
  }

  bf16_t* T0a = (bf16_t*)alloc((size_t)NV[0] * 32 * 2);
  bf16_t* T0b = (bf16_t*)alloc((size_t)NV[0] * 32 * 2);
  bf16_t* DN0 = (bf16_t*)alloc((size_t)NV[0] * 32 * 2);
  bf16_t* T1a = (bf16_t*)alloc((size_t)NV[1] * 64 * 2);
  bf16_t* T1b = (bf16_t*)alloc((size_t)NV[1] * 64 * 2);
  bf16_t* DN1 = (bf16_t*)alloc((size_t)NV[1] * 64 * 2);
  bf16_t* U1  = (bf16_t*)alloc((size_t)NV[1] * 32 * 2);
  bf16_t* T2a = (bf16_t*)alloc((size_t)NV[2] * 96 * 2);
  bf16_t* T2b = (bf16_t*)alloc((size_t)NV[2] * 96 * 2);
  bf16_t* DN2 = (bf16_t*)alloc((size_t)NV[2] * 96 * 2);
  bf16_t* U2  = (bf16_t*)alloc((size_t)NV[2] * 32 * 2);
  bf16_t* T3a = (bf16_t*)alloc((size_t)NV[3] * 128 * 2);
  bf16_t* T3b = (bf16_t*)alloc((size_t)NV[3] * 128 * 2);
  bf16_t* DN3 = (bf16_t*)alloc((size_t)NV[3] * 128 * 2);
  bf16_t* U3  = (bf16_t*)alloc((size_t)NV[3] * 32 * 2);
  if (off > ws_size) return;  // workspace too small; bail deterministically

  const int TB = 256;
  auto cdiv = [](long a, long b) -> int { return (int)((a + b - 1) / b); };

  const float* x = (const float*)d_in[0];
  const unsigned char* mask = (const unsigned char*)d_in[1];
  const float* Win = (const float*)d_in[2];

  // ---- masks & counts & guard page ----
  hipMemsetAsync(zp, 0, 512, stream);
  k_mask0<<<cdiv(NV[0], TB), TB, 0, stream>>>(mask, mf[0], NV[0]);
  for (int k = 0; k < 3; ++k)
    k_pool<<<cdiv(NV[k + 1], TB), TB, 0, stream>>>(mf[k], mf[k + 1], DIMS[k + 1], NV[k + 1]);
  hipMemsetAsync(cnt, 0, 4 * 4, stream);
  for (int k = 0; k < 4; ++k)
    k_count<<<cdiv(NV[k], TB), TB, 0, stream>>>(mf[k], cnt + k, NV[k]);

  // ---- weight repack (f32 -> WMMA B fragments, bf16) ----
  auto repack = [&](const float* src, bf16_t* dst, int T, int Cin, int Cout) {
    int total = (int)(wsz(T, Cin, Cout) / sizeof(bf16_t));
    k_repack<<<cdiv(total, TB), TB, 0, stream>>>(src, dst, T, Cin, Cout, total);
  };
  repack((const float*)d_in[5], wb[0], 27, 32, 32);
  repack((const float*)d_in[12], wb[1], 27, 64, 64);
  repack((const float*)d_in[19], wb[2], 27, 96, 96);
  repack((const float*)d_in[26], wb[3], 27, 128, 128);
  repack((const float*)d_in[9], wd[1], 8, 32, 64);
  repack((const float*)d_in[16], wd[2], 8, 64, 96);
  repack((const float*)d_in[23], wd[3], 8, 96, 128);
  repack((const float*)d_in[6], wsc[0], 1, 32, 32);
  repack((const float*)d_in[13], wsc[1], 1, 64, 32);
  repack((const float*)d_in[20], wsc[2], 1, 96, 32);
  repack((const float*)d_in[27], wsc[3], 1, 128, 32);
  repack((const float*)d_in[30], wu[0], 8, 32, 32);
  repack((const float*)d_in[31], wm[0], 27, 32, 32);
  repack((const float*)d_in[34], wu[1], 8, 32, 32);
  repack((const float*)d_in[35], wm[1], 27, 32, 32);
  repack((const float*)d_in[38], wu[2], 8, 32, 32);
  repack((const float*)d_in[39], wm[2], 27, 32, 32);

  auto bn = [&](const bf16_t* in, bf16_t* out, int gi, int bi, int lvl, int C) {
    hipMemsetAsync(stats, 0, 256 * 4, stream);
    k_bnstats<<<NV[lvl] / 64, TB, 0, stream>>>(in, mf[lvl], NV[lvl], C, stats);
    k_bnapply<<<cdiv((long)NV[lvl] * C, TB), TB, 0, stream>>>(
        in, out, mf[lvl], (const float*)d_in[gi], (const float*)d_in[bi], stats,
        cnt + lvl, NV[lvl], C);
  };

#define LAUNCH_SUBM(CIN, COUT, DIM, IN, OUT, WFR, LVL)                                   \
  k_subm3<CIN, COUT, DIM><<<cdiv((long)(NV[LVL] >> 4) * ((COUT) / 32), TB / 32), TB, 0,  \
                            stream>>>(IN, OUT, mf[LVL], WFR, zp)
#define LAUNCH_DOWN(CIN, COUT, DIMC, IN, OUT, WFR, LVLC)                                 \
  k_down<CIN, COUT, DIMC><<<cdiv((long)(NV[LVLC] >> 4) * ((COUT) / 32), TB / 32), TB, 0, \
                            stream>>>(IN, OUT, mf[LVLC], WFR)
#define LAUNCH_DEC(DIMC, IN, OUT, WFR, LVLC)                                             \
  k_deconv<DIMC><<<cdiv((long)(NV[LVLC] >> 4), TB / 32), TB, 0, stream>>>(               \
      IN, OUT, mf[(LVLC)-1], WFR)
#define LAUNCH_SC(CIN, IN, OUT, WFR, LVL, BETA)                                          \
  k_shortcut<CIN><<<cdiv((long)(NV[LVL] >> 4), TB / 32), TB, 0, stream>>>(               \
      IN, OUT, mf[LVL], WFR, NV[LVL], BETA)

  // ---- down path ----
  k_inconv<<<cdiv((long)NV[0] * 4, TB), TB, 0, stream>>>(x, mf[0], Win, T0a, DIMS[0], NV[0]);
  bn(T0a, T0b, 3, 4, 0, 32);                        // g0,b0
  LAUNCH_SUBM(32, 32, 64, T0b, DN0, wb[0], 0);      // downs[0]
  bn(DN0, T0b, 7, 8, 0, 32);                        // gd1,bd1
  LAUNCH_DOWN(32, 64, 32, T0b, T1a, wd[1], 1);
  bn(T1a, T1b, 10, 11, 1, 64);                      // g1,b1
  LAUNCH_SUBM(64, 64, 32, T1b, DN1, wb[1], 1);      // downs[1]
  bn(DN1, T1b, 14, 15, 1, 64);                      // gd2,bd2
  LAUNCH_DOWN(64, 96, 16, T1b, T2a, wd[2], 2);
  bn(T2a, T2b, 17, 18, 2, 96);                      // g2,b2
  LAUNCH_SUBM(96, 96, 16, T2b, DN2, wb[2], 2);      // downs[2]
  bn(DN2, T2b, 21, 22, 2, 96);                      // gd3,bd3
  LAUNCH_DOWN(96, 128, 8, T2b, T3a, wd[3], 3);
  bn(T3a, T3b, 24, 25, 3, 128);                     // g3,b3
  LAUNCH_SUBM(128, 128, 8, T3b, DN3, wb[3], 3);     // downs[3]

  // ---- up path ----
  LAUNCH_SC(128, DN3, U3, wsc[3], 3, 0);            // ups[0]
  bn(U3, T3a, 28, 29, 3, 32);                       // gu0,bu0
  LAUNCH_DEC(8, T3a, T2a, wu[0], 3);                // -> L2 fine
  LAUNCH_SC(96, DN2, T2a, wsc[2], 2, 1);
  LAUNCH_SUBM(32, 32, 16, T2a, U2, wm[0], 2);       // ups[1]
  bn(U2, T2b, 32, 33, 2, 32);                       // gu1,bu1
  LAUNCH_DEC(16, T2b, T1a, wu[1], 2);               // -> L1 fine
  LAUNCH_SC(64, DN1, T1a, wsc[1], 1, 1);
  LAUNCH_SUBM(32, 32, 32, T1a, U1, wm[1], 1);       // ups[2]
  bn(U1, T1b, 36, 37, 1, 32);                       // gu2,bu2
  LAUNCH_DEC(32, T1b, T0a, wu[2], 1);               // -> L0 fine
  LAUNCH_SC(32, DN0, T0a, wsc[0], 0, 1);
  LAUNCH_SUBM(32, 32, 64, T0a, T0b, wm[2], 0);      // ups[3]

  // ---- exports (tuple order: ups[1], ups[2], ups[3]) ----
  float* outp = (float*)d_out;
  k_export<<<cdiv((long)NV[2] * 32, TB), TB, 0, stream>>>(U2, outp, DIMS[2], NV[2]);
  k_export<<<cdiv((long)NV[1] * 32, TB), TB, 0, stream>>>(U1, outp + 262144, DIMS[1], NV[1]);
  k_export<<<cdiv((long)NV[0] * 32, TB), TB, 0, stream>>>(T0b, outp + 262144 + 2097152,
                                                          DIMS[0], NV[0]);
#undef LAUNCH_SUBM
#undef LAUNCH_DOWN
#undef LAUNCH_DEC
#undef LAUNCH_SC
}
